// PerformerAttention_49555332661589
// MI455X (gfx1250) — matrix operand order repaired and checked by reading, not compiled, not run on hardware
//
#include <hip/hip_runtime.h>
#include <hip/hip_bf16.h>
#include <stdint.h>

// ---------------------------------------------------------------------------
// Performer (FAVOR+) attention for MI455X / gfx1250.
// All GEMMs run on v_wmma_f32_16x16x32_bf16 (wave32 WMMA), elementwise math
// (LayerNorm, exp, max, 1/denom) stays in f32 VALU.
// Block tile 256x64x32, 8 waves, each wave = 32x64 patch = 2x4 WMMA tiles.
// A-tile staging uses GLOBAL_LOAD_ASYNC_TO_LDS_B128 (ASYNCcnt) when the
// toolchain exposes the builtin; otherwise batched load->store fallback.
// ---------------------------------------------------------------------------

#define DEVI __device__ __forceinline__

typedef __attribute__((ext_vector_type(16))) __bf16 v16bf;
typedef __attribute__((ext_vector_type(8)))  float  v8f;
typedef int v4i __attribute__((vector_size(16)));   // GCC-style, matches builtin

#if defined(__has_builtin)
#if __has_builtin(__builtin_amdgcn_global_load_async_to_lds_b128) && \
    __has_builtin(__builtin_amdgcn_s_wait_asynccnt)
#define USE_ASYNC_LDS 1
#endif
#endif
#ifndef USE_ASYNC_LDS
#define USE_ASYNC_LDS 0
#endif

// problem dims
#define BB     4
#define NSEQ   4096
#define DMODEL 1024
#define NH     16
#define DHEAD  64
#define MFEAT  256
#define NB     (BB * NSEQ)       // 16384 token rows
#define BHCNT  (BB * NH)         // 64 (batch*heads)
#define BHN    (BHCNT * NSEQ)    // 262144 head-rows

DEVI unsigned short f2bf(float x) {
  unsigned int u = __builtin_bit_cast(unsigned int, x);
  unsigned int r = u + 0x7FFFu + ((u >> 16) & 1u);   // round-to-nearest-even
  return (unsigned short)(r >> 16);
}
DEVI float bf2f(unsigned short h) {
  unsigned int u = ((unsigned int)h) << 16;
  return __builtin_bit_cast(float, u);
}

union FragBF { v16bf v; unsigned int d[8]; };

// ---------------------------------------------------------------------------
// Tiled WMMA GEMM: C[M,N] = alpha * op(A) @ B  (+bias[col]) (*rowscale[row])
// A,B are bf16 (raw u16); C is f32 or bf16. ATRANS: A stored [K,M] (lda = M
// stride) and used transposed. Batched via blockIdx.z with outer/inner strides
// (zo = z/innerCnt -> batch, zi = z%innerCnt -> head).
// Block: 256 threads = 8 waves. Tile 256x64x32. Wave -> 32x64 (2x4 WMMA).
// ---------------------------------------------------------------------------
template <bool ATRANS, bool OUT_BF16, bool HAS_BIAS, bool HAS_RS>
__global__ __launch_bounds__(256) void gemm_wmma_bf16(
    const unsigned short* __restrict__ A, const unsigned short* __restrict__ B,
    void* __restrict__ Cv, const float* __restrict__ bias,
    const float* __restrict__ rowscale,
    int Mdim, int Ndim, int Kdim, int lda, int ldb, int ldc,
    long aso, long asi, long bso, long bsi, long cso, long csi,
    long rso, long rsi, int innerCnt, float alpha) {
  constexpr int BM = 256, BN = 64, BK = 32, PAD = 8;
  __shared__ __align__(16) unsigned short As[BM][BK + PAD];
  __shared__ __align__(16) unsigned short Bs[BN][BK + PAD];  // K-contiguous

  const int  z  = blockIdx.z;
  const long zo = z / innerCnt, zi = z % innerCnt;
  const unsigned short* Ab = A + zo * aso + zi * asi;
  const unsigned short* Bb = B + zo * bso + zi * bsi;
  const long coff = zo * cso + zi * csi;
  const float* rs = HAS_RS ? (rowscale + zo * rso + zi * rsi) : nullptr;

  const int blockM = blockIdx.y * BM;
  const int blockN = blockIdx.x * BN;
  const int tid  = threadIdx.x;
  const int lane = tid & 31;
  const int wid  = tid >> 5;
  const int wM   = wid * 32;         // 8 waves along M (256 rows)
  const int half = lane >> 4;        // lane half (0/1)
  const int l15  = lane & 15;

  v8f acc[2][4] = {};

  for (int k0 = 0; k0 < Kdim; k0 += BK) {
    // ---- stage A tile (256 x 32) ----
    if constexpr (!ATRANS) {
      const int r  = tid >> 2;            // 0..63
      const int cg = (tid & 3) * 8;       // 0/8/16/24
#if USE_ASYNC_LDS
#pragma unroll
      for (int p = 0; p < 4; ++p) {
        const int row = r + p * 64;
        const unsigned short* src = Ab + (long)(blockM + row) * lda + k0 + cg;
        __builtin_amdgcn_global_load_async_to_lds_b128(
            (__attribute__((address_space(1))) v4i*)src,
            (__attribute__((address_space(3))) v4i*)&As[row][cg], 0, 0);
      }
#else
      uint4 av[4];
#pragma unroll
      for (int p = 0; p < 4; ++p) {
        const int row = r + p * 64;
        av[p] = *reinterpret_cast<const uint4*>(
            Ab + (long)(blockM + row) * lda + k0 + cg);
      }
#pragma unroll
      for (int p = 0; p < 4; ++p)
        *reinterpret_cast<uint4*>(&As[r + p * 64][cg]) = av[p];
#endif
    } else {
      const int kk = tid >> 5;            // 0..7
      const int mg = (tid & 31) * 8;      // 0..248
      uint4 av[4];
#pragma unroll
      for (int p = 0; p < 4; ++p) {
        const int krow = kk + p * 8;
        av[p] = *reinterpret_cast<const uint4*>(
            Ab + (long)(k0 + krow) * lda + blockM + mg);
      }
#pragma unroll
      for (int p = 0; p < 4; ++p) {
        const int krow = kk + p * 8;
        __align__(16) unsigned short tmp[8];
        *reinterpret_cast<uint4*>(tmp) = av[p];
#pragma unroll
        for (int e = 0; e < 8; ++e) As[mg + e][krow] = tmp[e];
      }
    }
    // ---- stage B tile (32 x 64), stored transposed: Bs[col][k] ----
    {
      const int kk = tid >> 3;            // 0..31
      const int cg = (tid & 7) * 8;       // 0..56
      const unsigned short* src = Bb + (long)(k0 + kk) * ldb + blockN + cg;
      uint4 dv = *reinterpret_cast<const uint4*>(src);
      __align__(16) unsigned short tmp[8];
      *reinterpret_cast<uint4*>(tmp) = dv;
#pragma unroll
      for (int e = 0; e < 8; ++e) Bs[cg + e][kk] = tmp[e];
    }
    // prefetch next K tiles into near caches while we compute
    if (k0 + BK < Kdim) {
      if constexpr (!ATRANS) {
        // one thread per tile row: a single cacheline covers the whole BK chunk
        __builtin_prefetch(Ab + (long)(blockM + tid) * lda + (k0 + BK), 0, 3);
      } else {
        __builtin_prefetch(Ab + (long)(k0 + BK + (tid >> 3)) * lda + blockM +
                               (tid & 7) * 32, 0, 3);
      }
      __builtin_prefetch(Bb + (long)(k0 + BK + (tid >> 3)) * ldb + blockN +
                             (tid & 7) * 8, 0, 3);
    }
#if USE_ASYNC_LDS
    if constexpr (!ATRANS) __builtin_amdgcn_s_wait_asynccnt(0);
#endif
    __syncthreads();

    // ---- build fragments per ISA 7.12.2 16-bit layouts ----
    FragBF afr[2], bfr[4];
#pragma unroll
    for (int mi = 0; mi < 2; ++mi) {
      const int arow = wM + mi * 16 + l15;    // M = l15
#pragma unroll
      for (int v = 0; v < 4; ++v) {
        // VGPR v   : K = half*8 + 2v, +1 ; VGPR 4+v : K = 16 + half*8 + 2v, +1
        afr[mi].d[v]     = *reinterpret_cast<const unsigned int*>(
            &As[arow][half * 8 + 2 * v]);
        afr[mi].d[4 + v] = *reinterpret_cast<const unsigned int*>(
            &As[arow][16 + half * 8 + 2 * v]);
      }
    }
#pragma unroll
    for (int ni = 0; ni < 4; ++ni) {
      const int bcol = ni * 16 + l15;         // N = l15
#pragma unroll
      for (int v = 0; v < 8; ++v) {
        // VGPR v<4: K = half*8 + 2v, +1 ; VGPR v>=4: K = 16 + half*8 + 2(v-4), +1
        bfr[ni].d[v] = *reinterpret_cast<const unsigned int*>(
            &Bs[bcol][(v >> 2) * 16 + half * 8 + 2 * (v & 3)]);
      }
    }
#pragma unroll
    for (int mi = 0; mi < 2; ++mi)
#pragma unroll
      for (int ni = 0; ni < 4; ++ni)
        acc[mi][ni] = __builtin_amdgcn_wmma_f32_16x16x32_bf16(
            false, afr[mi].v, false, bfr[ni].v, (short)0, acc[mi][ni],
            false, false);

    __syncthreads();
  }

  // ---- epilogue: C VGPR v -> row = v + half*8, col = l15 ----
#pragma unroll
  for (int mi = 0; mi < 2; ++mi)
#pragma unroll
    for (int ni = 0; ni < 4; ++ni)
#pragma unroll
      for (int v = 0; v < 8; ++v) {
        const int row = blockM + wM + mi * 16 + half * 8 + v;
        const int col = blockN + ni * 16 + l15;
        float val = acc[mi][ni][v] * alpha;
        if constexpr (HAS_RS)   val *= rs[row];
        if constexpr (HAS_BIAS) val += bias[col];
        const long cidx = coff + (long)row * ldc + col;
        if constexpr (OUT_BF16)
          ((unsigned short*)Cv)[cidx] = f2bf(val);
        else
          ((float*)Cv)[cidx] = val;
      }
}

// ---------------------------------------------------------------------------
// Elementwise / reduction helpers (f32 VALU)
// ---------------------------------------------------------------------------
__global__ void cvt_f32_bf16(const float* __restrict__ in,
                             unsigned short* __restrict__ out, long n) {
  long i = (long)blockIdx.x * blockDim.x + threadIdx.x;
  if (i < n) out[i] = f2bf(in[i]);
}

// projT[d][m] = normalizer * proj[m][d]  (fold DH^-0.25 into the projection)
__global__ void projT_scale(const float* __restrict__ proj,
                            unsigned short* __restrict__ pT, float s) {
  int i = blockIdx.x * 256 + threadIdx.x;  // over MFEAT*DHEAD
  int m = i >> 6, d = i & 63;
  pT[d * MFEAT + m] = f2bf(proj[i] * s);
}

__global__ __launch_bounds__(256) void layernorm_bf16(
    const float* __restrict__ x, const float* __restrict__ g,
    const float* __restrict__ beta, unsigned short* __restrict__ xn) {
  const int row = blockIdx.x, t = threadIdx.x;
  const float* xr = x + (long)row * DMODEL;
  float s = 0.f, ss = 0.f;
  for (int i = t; i < DMODEL; i += 256) { float v = xr[i]; s += v; ss += v * v; }
  __shared__ float s1[256], s2[256];
  s1[t] = s; s2[t] = ss; __syncthreads();
  for (int o = 128; o > 0; o >>= 1) {
    if (t < o) { s1[t] += s1[t + o]; s2[t] += s2[t + o]; }
    __syncthreads();
  }
  const float mean = s1[0] * (1.0f / DMODEL);
  const float var  = s2[0] * (1.0f / DMODEL) - mean * mean;
  const float inv  = rsqrtf(var + 1e-5f);
  for (int i = t; i < DMODEL; i += 256)
    xn[(long)row * DMODEL + i] = f2bf((xr[i] - mean) * inv * g[i] + beta[i]);
}

// diag[b,h,n] = coef * sum_d q[b,n,h*64+d]^2   (coef = 0.5 * DH^-0.5)
__global__ void head_sumsq(const unsigned short* __restrict__ t,
                           float* __restrict__ diag, float coef) {
  long i = (long)blockIdx.x * 256 + threadIdx.x;  // over BHN
  int n  = (int)(i & (NSEQ - 1));
  int bh = (int)(i >> 12);
  int h = bh & (NH - 1), b = bh >> 4;
  const unsigned short* r = t + ((long)(b * NSEQ + n)) * DMODEL + h * DHEAD;
  float s = 0.f;
#pragma unroll
  for (int d = 0; d < DHEAD; ++d) { float v = bf2f(r[d]); s += v * v; }
  diag[i] = s * coef;
}

// per-row max + exp for queries: qp = ratio*(exp(dash - diag - rowmax) + eps)
__global__ __launch_bounds__(256) void q_featuremap(
    const float* __restrict__ dash, const float* __restrict__ diag,
    unsigned short* __restrict__ qp) {
  const long row = blockIdx.x;
  const int  t   = threadIdx.x;     // 256 == MFEAT
  float v = dash[row * MFEAT + t];
  __shared__ float sh[256];
  sh[t] = v; __syncthreads();
  for (int o = 128; o > 0; o >>= 1) {
    if (t < o) sh[t] = fmaxf(sh[t], sh[t + o]);
    __syncthreads();
  }
  float val = 0.0625f * (__expf(v - diag[row] - sh[0]) + 1e-4f);
  qp[row * MFEAT + t] = f2bf(val);
}

// global max over (n,m) per (b,h)
__global__ __launch_bounds__(256) void bh_max(const float* __restrict__ dash,
                                              float* __restrict__ stab) {
  const int bh = blockIdx.x, t = threadIdx.x;
  const float* p = dash + (long)bh * NSEQ * MFEAT;
  float m = -3.4e38f;
  for (long i = t; i < (long)NSEQ * MFEAT; i += 256) m = fmaxf(m, p[i]);
  __shared__ float sh[256];
  sh[t] = m; __syncthreads();
  for (int o = 128; o > 0; o >>= 1) {
    if (t < o) sh[t] = fmaxf(sh[t], sh[t + o]);
    __syncthreads();
  }
  if (t == 0) stab[bh] = sh[0];
}

__global__ void k_featuremap(const float* __restrict__ dash,
                             const float* __restrict__ diag,
                             const float* __restrict__ stab,
                             unsigned short* __restrict__ kp) {
  long i   = (long)blockIdx.x * 256 + threadIdx.x;
  long row = i >> 8;     // / MFEAT
  long bh  = row >> 12;  // / NSEQ
  float val = 0.0625f * (__expf(dash[i] - diag[row] - stab[bh]) + 1e-4f);
  kp[i] = f2bf(val);
}

// k_sum[b,h,m] = sum_n kp[b,h,n,m]
__global__ __launch_bounds__(256) void k_colsum(
    const unsigned short* __restrict__ kp, float* __restrict__ ksum) {
  const int bh = blockIdx.x, m = threadIdx.x;
  const unsigned short* p = kp + (long)bh * NSEQ * MFEAT + m;
  float s = 0.f;
  for (int n = 0; n < NSEQ; ++n) s += bf2f(p[(long)n * MFEAT]);
  ksum[bh * MFEAT + m] = s;
}

// d_inv[b,h,n] = 1 / dot(qp[b,h,n,:], ksum[b,h,:])
__global__ __launch_bounds__(256) void d_inverse(
    const unsigned short* __restrict__ qp, const float* __restrict__ ksum,
    float* __restrict__ dinv) {
  const long row = blockIdx.x;
  const int  t   = threadIdx.x;
  const long bh  = row >> 12;
  float v = bf2f(qp[row * MFEAT + t]) * ksum[bh * MFEAT + t];
  __shared__ float sh[256];
  sh[t] = v; __syncthreads();
  for (int o = 128; o > 0; o >>= 1) {
    if (t < o) sh[t] += sh[t + o];
    __syncthreads();
  }
  if (t == 0) dinv[row] = 1.0f / sh[0];
}

// ---------------------------------------------------------------------------
extern "C" void kernel_launch(void* const* d_in, const int* in_sizes, int n_in,
                              void* d_out, int out_size, void* d_ws,
                              size_t ws_size, hipStream_t stream) {
  (void)in_sizes; (void)n_in; (void)out_size; (void)ws_size;
  const float* x    = (const float*)d_in[0];
  const float* lng  = (const float*)d_in[1];
  const float* lnb  = (const float*)d_in[2];
  const float* Wq   = (const float*)d_in[3];
  const float* Wk   = (const float*)d_in[4];
  const float* Wv   = (const float*)d_in[5];
  const float* Wo   = (const float*)d_in[6];
  const float* bo   = (const float*)d_in[7];
  const float* proj = (const float*)d_in[8];
  float* out = (float*)d_out;

  char* ws = (char*)d_ws;
  size_t off = 0;
  auto alloc = [&](size_t bytes) -> char* {
    char* p = ws + off;
    off += (bytes + 255) & ~(size_t)255;
    return p;
  };

  unsigned short* xn_bf  = (unsigned short*)alloc((size_t)NB * DMODEL * 2);
  unsigned short* wq_bf  = (unsigned short*)alloc((size_t)DMODEL * DMODEL * 2);
  unsigned short* wk_bf  = (unsigned short*)alloc((size_t)DMODEL * DMODEL * 2);
  unsigned short* wv_bf  = (unsigned short*)alloc((size_t)DMODEL * DMODEL * 2);
  unsigned short* wo_bf  = (unsigned short*)alloc((size_t)DMODEL * DMODEL * 2);
  unsigned short* pT_bf  = (unsigned short*)alloc((size_t)DHEAD * MFEAT * 2);
  unsigned short* q_bf   = (unsigned short*)alloc((size_t)NB * DMODEL * 2);
  unsigned short* k_bf   = (unsigned short*)alloc((size_t)NB * DMODEL * 2);
  unsigned short* v_bf   = (unsigned short*)alloc((size_t)NB * DMODEL * 2);
  float*          diag_q = (float*)alloc((size_t)BHN * 4);
  float*          diag_k = (float*)alloc((size_t)BHN * 4);
  float*          dash   = (float*)alloc((size_t)BHN * MFEAT * 4);  // reused q->k
  unsigned short* qp_bf  = (unsigned short*)alloc((size_t)BHN * MFEAT * 2);
  unsigned short* kp_bf  = (unsigned short*)alloc((size_t)BHN * MFEAT * 2);
  float*          stab_k = (float*)alloc((size_t)BHCNT * 4);
  float*          ksum   = (float*)alloc((size_t)BHCNT * MFEAT * 4);
  float*          dinv   = (float*)alloc((size_t)BHN * 4);
  float*          ctx_f  = (float*)alloc((size_t)BHCNT * MFEAT * DHEAD * 4);
  unsigned short* ctx_bf = (unsigned short*)alloc((size_t)BHCNT * MFEAT * DHEAD * 2);
  unsigned short* at_bf  = (unsigned short*)alloc((size_t)NB * DMODEL * 2);

  const dim3 blk(256);
  const long WSZ = (long)DMODEL * DMODEL;

  // weight / projection conversion to bf16
  cvt_f32_bf16<<<WSZ / 256, blk, 0, stream>>>(Wq, wq_bf, WSZ);
  cvt_f32_bf16<<<WSZ / 256, blk, 0, stream>>>(Wk, wk_bf, WSZ);
  cvt_f32_bf16<<<WSZ / 256, blk, 0, stream>>>(Wv, wv_bf, WSZ);
  cvt_f32_bf16<<<WSZ / 256, blk, 0, stream>>>(Wo, wo_bf, WSZ);
  projT_scale<<<(MFEAT * DHEAD) / 256, blk, 0, stream>>>(
      proj, pT_bf, 0.35355339059327373f /* DH^-0.25 */);

  // LayerNorm -> bf16
  layernorm_bf16<<<NB, blk, 0, stream>>>(x, lng, lnb, xn_bf);

  // Q/K/V projections (q pre-scaled by DH^-0.5)
  {
    dim3 g(DMODEL / 64, NB / 256, 1);
    gemm_wmma_bf16<false, true, false, false><<<g, blk, 0, stream>>>(
        xn_bf, wq_bf, q_bf, nullptr, nullptr, NB, DMODEL, DMODEL, DMODEL,
        DMODEL, DMODEL, 0, 0, 0, 0, 0, 0, 0, 0, 1, 0.125f);
    gemm_wmma_bf16<false, true, false, false><<<g, blk, 0, stream>>>(
        xn_bf, wk_bf, k_bf, nullptr, nullptr, NB, DMODEL, DMODEL, DMODEL,
        DMODEL, DMODEL, 0, 0, 0, 0, 0, 0, 0, 0, 1, 1.0f);
    gemm_wmma_bf16<false, true, false, false><<<g, blk, 0, stream>>>(
        xn_bf, wv_bf, v_bf, nullptr, nullptr, NB, DMODEL, DMODEL, DMODEL,
        DMODEL, DMODEL, 0, 0, 0, 0, 0, 0, 0, 0, 1, 1.0f);
  }

  // diag terms (0.5 * DH^-0.5 = 0.0625)
  head_sumsq<<<BHN / 256, blk, 0, stream>>>(q_bf, diag_q, 0.0625f);
  head_sumsq<<<BHN / 256, blk, 0, stream>>>(k_bf, diag_k, 0.0625f);

  const long S_DASH = (long)NSEQ * MFEAT;

  // query features: dash = q_head @ projT -> qp (row-max stabilized exp)
  {
    dim3 g(MFEAT / 64, NSEQ / 256, BHCNT);
    gemm_wmma_bf16<false, false, false, false><<<g, blk, 0, stream>>>(
        q_bf, pT_bf, dash, nullptr, nullptr, NSEQ, MFEAT, DHEAD, DMODEL, MFEAT,
        MFEAT, (long)NSEQ * DMODEL, DHEAD, 0, 0, (long)NH * S_DASH, S_DASH,
        0, 0, NH, 1.0f);
  }
  q_featuremap<<<BHN, blk, 0, stream>>>(dash, diag_q, qp_bf);

  // key features: reuse dash buffer, global (per b,h) stabilization
  {
    dim3 g(MFEAT / 64, NSEQ / 256, BHCNT);
    gemm_wmma_bf16<false, false, false, false><<<g, blk, 0, stream>>>(
        k_bf, pT_bf, dash, nullptr, nullptr, NSEQ, MFEAT, DHEAD, DMODEL, MFEAT,
        MFEAT, (long)NSEQ * DMODEL, DHEAD, 0, 0, (long)NH * S_DASH, S_DASH,
        0, 0, NH, 1.0f);
  }
  bh_max<<<BHCNT, blk, 0, stream>>>(dash, stab_k);
  k_featuremap<<<(int)(((long)BHN * MFEAT) / 256), blk, 0, stream>>>(
      dash, diag_k, stab_k, kp_bf);

  // denominator pieces
  k_colsum<<<BHCNT, blk, 0, stream>>>(kp_bf, ksum);
  d_inverse<<<BHN, blk, 0, stream>>>(qp_bf, ksum, dinv);

  // context[b,h] = kp^T @ v_head : [256 x 64], K = 4096
  {
    dim3 g(DHEAD / 64, MFEAT / 256, BHCNT);
    gemm_wmma_bf16<true, false, false, false><<<g, blk, 0, stream>>>(
        kp_bf, v_bf, ctx_f, nullptr, nullptr, MFEAT, DHEAD, NSEQ, MFEAT,
        DMODEL, DHEAD, (long)NH * S_DASH, S_DASH, (long)NSEQ * DMODEL, DHEAD,
        (long)NH * MFEAT * DHEAD, (long)MFEAT * DHEAD, 0, 0, NH, 1.0f);
  }
  cvt_f32_bf16<<<(int)(((long)BHCNT * MFEAT * DHEAD) / 256), blk, 0, stream>>>(
      ctx_f, ctx_bf, (long)BHCNT * MFEAT * DHEAD);

  // out_head = (qp @ context) * d_inv, scattered into [B*N, H*DH] bf16
  {
    dim3 g(DHEAD / 64, NSEQ / 256, BHCNT);
    gemm_wmma_bf16<false, true, false, true><<<g, blk, 0, stream>>>(
        qp_bf, ctx_bf, at_bf, nullptr, dinv, NSEQ, DHEAD, MFEAT, MFEAT, DHEAD,
        DMODEL, (long)NH * S_DASH, S_DASH, (long)NH * MFEAT * DHEAD,
        (long)MFEAT * DHEAD, (long)NSEQ * DMODEL, DHEAD, (long)NH * NSEQ,
        (long)NSEQ, NH, 1.0f);
  }

  // final projection + bias -> f32 output
  {
    dim3 g(DMODEL / 64, NB / 256, 1);
    gemm_wmma_bf16<false, false, true, false><<<g, blk, 0, stream>>>(
        at_bf, wo_bf, out, bo, nullptr, NB, DMODEL, DMODEL, DMODEL, DMODEL,
        DMODEL, 0, 0, 0, 0, 0, 0, 0, 0, 1, 1.0f);
  }
}
